// Int8Linear_41644002902329
// MI455X (gfx1250) — 8-bit matrix operand order repaired and checked by reading, compiled, not run on hardware
//
#include <hip/hip_runtime.h>
#include <hip/hip_bf16.h>

typedef __attribute__((ext_vector_type(8))) int v8i;
typedef __attribute__((ext_vector_type(4))) int v4i;
typedef __attribute__((ext_vector_type(2))) int v2i;

#define QEPS 1e-5f

__global__ void int8lin_zero_kernel(unsigned int* __restrict__ maxbits) {
    if (threadIdx.x == 0) maxbits[0] = 0u;
}

__global__ __launch_bounds__(256) void int8lin_absmax_kernel(
    const float4* __restrict__ x4, long n4, unsigned int* __restrict__ maxbits) {
    long tid    = (long)blockIdx.x * blockDim.x + threadIdx.x;
    long stride = (long)gridDim.x * blockDim.x;
    float m = 0.0f;
    for (long i = tid; i < n4; i += stride) {
        float4 v = x4[i];
        m = fmaxf(m, fmaxf(fmaxf(fabsf(v.x), fabsf(v.y)),
                           fmaxf(fabsf(v.z), fabsf(v.w))));
    }
#pragma unroll
    for (int off = 16; off > 0; off >>= 1)
        m = fmaxf(m, __shfl_xor(m, off, 32));
    if ((threadIdx.x & 31) == 0)
        atomicMax(maxbits, __float_as_uint(m));
}

__global__ __launch_bounds__(256) void int8lin_quant_kernel(
    const float4* __restrict__ x4, int* __restrict__ xq, long n4,
    const unsigned int* __restrict__ maxbits) {
    long i = (long)blockIdx.x * blockDim.x + threadIdx.x;
    if (i >= n4) return;
    float scale = fmaxf(__uint_as_float(*maxbits) * (1.0f / 127.0f), QEPS);
    float inv   = 1.0f / scale;
    float4 v = x4[i];
    int a = __float2int_rn(v.x * inv); a = max(-128, min(127, a));
    int b = __float2int_rn(v.y * inv); b = max(-128, min(127, b));
    int c = __float2int_rn(v.z * inv); c = max(-128, min(127, c));
    int d = __float2int_rn(v.w * inv); d = max(-128, min(127, d));
    xq[i] = (a & 0xff) | ((b & 0xff) << 8) | ((c & 0xff) << 16) | (d << 24);
}


__device__ __forceinline__ v8i int8lin_load_a(const signed char* p) {
    v2i a0 = *(const v2i*)(p);
    v2i a1 = *(const v2i*)(p + 8);
    v2i a2 = *(const v2i*)(p + 32);
    v2i a3 = *(const v2i*)(p + 40);
    v8i r;
    r[0] = a0[0]; r[1] = a0[1];
    r[2] = a1[0]; r[3] = a1[1];
    r[4] = a2[0]; r[5] = a2[1];
    r[6] = a3[0]; r[7] = a3[1];
    return r;
}

__device__ __forceinline__ v8i int8lin_load_b(const signed char* p) {
    v4i lo = *(const v4i*)(p);
    v4i hi = *(const v4i*)(p + 32);
    v8i r;
    r[0] = lo[0]; r[1] = lo[1]; r[2] = lo[2]; r[3] = lo[3];
    r[4] = hi[0]; r[5] = hi[1]; r[6] = hi[2]; r[7] = hi[3];
    return r;
}

__device__ __forceinline__ void int8lin_epilogue(
    const v8i& c, int mTop, int n, int half, float s, float bb,
    float* __restrict__ out, int N) {
#pragma unroll
    for (int r = 0; r < 8; ++r) {
        size_t idx = (size_t)(mTop + (half << 3) + r) * (size_t)N + (size_t)n;
        __builtin_nontemporal_store((float)c[r] * s + bb, out + idx);
    }
}

__global__ __launch_bounds__(256) void int8lin_gemm_kernel(
    const signed char* __restrict__ Aq,
    const signed char* __restrict__ W,
    const float* __restrict__ wscale,
    const float* __restrict__ bias,
    const unsigned int* __restrict__ maxbits,
    float* __restrict__ out,
    int M, int N, int K) {
    const int lane = threadIdx.x & 31;
    const int wave = threadIdx.x >> 5;
    const int half = lane >> 4;
    const int l16  = lane & 15;

    const int mBase = (blockIdx.y * 8 + wave) * 32;
    const int nBase = blockIdx.x * 64;

    const signed char* aRow0 = Aq + (size_t)(mBase + l16) * K + half * 16;
    const signed char* aRow1 = aRow0 + (size_t)16 * K;
    const signed char* bCol0 = W + (size_t)(nBase + l16) * K + half * 16;
    const signed char* bCol1 = bCol0 + (size_t)16 * K;
    const signed char* bCol2 = bCol0 + (size_t)32 * K;
    const signed char* bCol3 = bCol0 + (size_t)48 * K;

    v8i c00 = {}, c01 = {}, c02 = {}, c03 = {};
    v8i c10 = {}, c11 = {}, c12 = {}, c13 = {};

    for (int k = 0; k < K; k += 64) {
        v8i a0 = int8lin_load_a(aRow0 + k);
        v8i a1 = int8lin_load_a(aRow1 + k);
        v8i b0 = int8lin_load_b(bCol0 + k);
        v8i b1 = int8lin_load_b(bCol1 + k);
        v8i b2 = int8lin_load_b(bCol2 + k);
        v8i b3 = int8lin_load_b(bCol3 + k);
        c00 = __builtin_amdgcn_wmma_i32_16x16x64_iu8(true, a0, true, b0, c00, false, false);
        c01 = __builtin_amdgcn_wmma_i32_16x16x64_iu8(true, a0, true, b1, c01, false, false);
        c02 = __builtin_amdgcn_wmma_i32_16x16x64_iu8(true, a0, true, b2, c02, false, false);
        c03 = __builtin_amdgcn_wmma_i32_16x16x64_iu8(true, a0, true, b3, c03, false, false);
        c10 = __builtin_amdgcn_wmma_i32_16x16x64_iu8(true, a1, true, b0, c10, false, false);
        c11 = __builtin_amdgcn_wmma_i32_16x16x64_iu8(true, a1, true, b1, c11, false, false);
        c12 = __builtin_amdgcn_wmma_i32_16x16x64_iu8(true, a1, true, b2, c12, false, false);
        c13 = __builtin_amdgcn_wmma_i32_16x16x64_iu8(true, a1, true, b3, c13, false, false);
    }

    const float xs = fmaxf(__uint_as_float(*maxbits) * (1.0f / 127.0f), QEPS);
#pragma unroll
    for (int ni = 0; ni < 4; ++ni) {
        const int n = nBase + ni * 16 + l16;
        const float s  = xs * wscale[n];
        const float bb = bias[n];
        const v8i& cLo = (ni == 0) ? c00 : (ni == 1) ? c01 : (ni == 2) ? c02 : c03;
        const v8i& cHi = (ni == 0) ? c10 : (ni == 1) ? c11 : (ni == 2) ? c12 : c13;
        int8lin_epilogue(cLo, mBase,      n, half, s, bb, out, N);
        int8lin_epilogue(cHi, mBase + 16, n, half, s, bb, out, N);
    }
}

extern "C" void kernel_launch(void* const* d_in, const int* in_sizes, int n_in,
                              void* d_out, int out_size, void* d_ws, size_t ws_size,
                              hipStream_t stream) {
    const float*       x      = (const float*)d_in[0];
    const signed char* w8     = (const signed char*)d_in[1];
    const float*       wscale = (const float*)d_in[2];
    const float*       bias   = (const float*)d_in[3];
    float*             out    = (float*)d_out;

    const int  N  = in_sizes[2];
    const int  K  = in_sizes[1] / N;
    const long nx = (long)in_sizes[0];
    const int  M  = (int)(nx / (long)K);

    unsigned int* maxbits = (unsigned int*)d_ws;
    signed char*  xq      = (signed char*)d_ws + 256;

    int8lin_zero_kernel<<<1, 32, 0, stream>>>(maxbits);

    const long n4 = nx >> 2;
    int8lin_absmax_kernel<<<2048, 256, 0, stream>>>((const float4*)x, n4, maxbits);

    const int qblocks = (int)((n4 + 255) / 256);
    int8lin_quant_kernel<<<qblocks, 256, 0, stream>>>((const float4*)x, (int*)xq, n4, maxbits);

    dim3 grid(N / 64, M / 256);
    int8lin_gemm_kernel<<<grid, 256, 0, stream>>>(xq, w8, wscale, bias, maxbits,
                                                  out, M, N, K);
}
